// SpatioTemporalUnit_39161511804964
// MI455X (gfx1250) — matrix operand order repaired and checked by reading, not compiled, not run on hardware
//
#include <hip/hip_runtime.h>
#include <hip/hip_bf16.h>
#include <math.h>

// ---------------- problem dimensions ----------------
#define B_    64
#define C_    256
#define T_    64
#define V_    25
#define S_    3
#define H_    4
#define REL_  32
#define HID_  64
#define HOP1  5          // MAXHOP+1
#define HD_   64         // C_/H_
#define TV    1600       // T_*V_
#define VV    625        // V_*V_
#define NT    102400     // B_*T_*V_  (tokens)
#define NOUT  26214400   // B_*C_*T_*V_

typedef __attribute__((ext_vector_type(16))) _Float16 v16h;
typedef __attribute__((ext_vector_type(8)))  _Float16 v8h;
typedef __attribute__((ext_vector_type(8)))  float    v8f;

// 16-byte async global->LDS copy (ASYNCcnt tracked). dst must be 16B-aligned LDS,
// src 16B-aligned global. LDS address = low 32 bits of the generic pointer.
__device__ __forceinline__ void async_copy16(void* lds_dst, const void* gsrc) {
  unsigned d = (unsigned)(unsigned long long)(uintptr_t)lds_dst;
  unsigned long long s = (unsigned long long)(uintptr_t)gsrc;
  asm volatile("global_load_async_to_lds_b128 %0, %1, off" :: "v"(d), "v"(s) : "memory");
}
__device__ __forceinline__ void async_wait0() {
  asm volatile("s_wait_asynccnt 0x0" ::: "memory");
}

// ---------------- generic f16-WMMA linear: Out[m,n] = sum_k A[m,k]*B(k,n) ----------------
// A: row-major [m][k], lda (multiple of 4).
// B: b_k_major=1 -> element(k,n) at Bm[k*ldb+n]  (x3: x[b] is [c][tv])
//    b_k_major=0 -> element(k,n) at Bm[n*ldb+k]  (linear-layer weights [n][k])
// Epilogue: optional bias (per-m or per-n), optional residual gathered from x[B,C,T,V],
//           act: 0=none 1=relu 2=sigmoid.
// batch_mode 1: grid.z = b*S_+s ; A += s*aStride, Bm += b*bStride, Out += z*oStride, bias += s*biasStride.
// Tile: 128 x TBN x 32, 256 threads (8 waves, 4x2 wave grid).

#define APITCH 40   // halves; 80B rows (16B multiple)
#define BPITCH 40

template<int TBN>
__global__ void __launch_bounds__(256) k_wmma_linear(
    const float* __restrict__ A, long lda,
    const float* __restrict__ Bm, long ldb, int b_k_major,
    const float* __restrict__ bias, int bias_per_m,
    const float* __restrict__ xres,
    float* __restrict__ Out, long ldo,
    int M, int N, int K, int act,
    int batch_mode, long aStride, long bStride, long oStride, long biasStride)
{
  constexpr int NI = TBN / 32;              // N subtiles per wave
  constexpr int AV = (128 * 32) / (256 * 4);  // float4 per thread for A (=4)
  constexpr int BV = (TBN * 32) / (256 * 4);  // float4 per thread for B (2 or 4)
  __shared__ __align__(16) _Float16 As[128 * APITCH];
  __shared__ __align__(16) _Float16 Bs[TBN * BPITCH];

  if (batch_mode == 1) {
    int z = blockIdx.z;
    int s = z % S_, b = z / S_;
    A   += (long)s * aStride;
    Bm  += (long)b * bStride;
    Out += (long)z * oStride;
    if (bias) bias += (long)s * biasStride;
  }

  const int m0  = blockIdx.x * 128;
  const int n0  = blockIdx.y * TBN;
  const int tid = threadIdx.x;
  const int lane = tid & 31;
  const int w    = tid >> 5;
  const int wm   = w >> 1;   // 0..3 -> 32-row slice
  const int wn   = w & 1;    // 0..1 -> TBN/2-col slice

  v8f acc[2][NI];
  const v8f vzero = {0.f,0.f,0.f,0.f,0.f,0.f,0.f,0.f};
  #pragma unroll
  for (int mi = 0; mi < 2; mi++)
    #pragma unroll
    for (int ni = 0; ni < NI; ni++) acc[mi][ni] = vzero;

  for (int k0 = 0; k0 < K; k0 += 32) {
    // ---- phase 0: gather tiles into registers (loads pipeline, single wait) ----
    float4 aR[AV];
    #pragma unroll
    for (int it = 0; it < AV; it++) {
      int idx = tid + it * 256;               // 0..1023
      int k4 = idx & 7, m = idx >> 3;
      aR[it] = *(const float4*)(A + (long)(m0 + m) * lda + (k0 + (k4 << 2)));
    }
    float4 bR[BV];
    if (b_k_major) {
      #pragma unroll
      for (int it = 0; it < BV; it++) {
        int idx = tid + it * 256;
        int n4 = idx & (TBN/4 - 1), k = idx / (TBN/4);
        bR[it] = *(const float4*)(Bm + (long)(k0 + k) * ldb + (n0 + (n4 << 2)));
      }
    } else {
      #pragma unroll
      for (int it = 0; it < BV; it++) {
        int idx = tid + it * 256;
        int k4 = idx & 7, n = idx >> 3;
        bR[it] = *(const float4*)(Bm + (long)(n0 + n) * ldb + (k0 + (k4 << 2)));
      }
    }
    if (k0 + 32 < K)
      __builtin_prefetch(&A[(long)(m0 + (tid >> 1)) * lda + (k0 + 32) + ((tid & 1) << 4)], 0, 0);

    __syncthreads();   // protect previous iteration's LDS reads
    // ---- phase 1: convert to f16, store to LDS (packed 8B stores) ----
    #pragma unroll
    for (int it = 0; it < AV; it++) {
      int idx = tid + it * 256;
      int k4 = idx & 7, m = idx >> 3;
      union { _Float16 h[4]; unsigned long long q; } u;
      u.h[0] = (_Float16)aR[it].x; u.h[1] = (_Float16)aR[it].y;
      u.h[2] = (_Float16)aR[it].z; u.h[3] = (_Float16)aR[it].w;
      *(unsigned long long*)(As + m * APITCH + (k4 << 2)) = u.q;
    }
    if (b_k_major) {
      #pragma unroll
      for (int it = 0; it < BV; it++) {
        int idx = tid + it * 256;
        int n4 = idx & (TBN/4 - 1), k = idx / (TBN/4);
        Bs[(n4*4+0) * BPITCH + k] = (_Float16)bR[it].x;
        Bs[(n4*4+1) * BPITCH + k] = (_Float16)bR[it].y;
        Bs[(n4*4+2) * BPITCH + k] = (_Float16)bR[it].z;
        Bs[(n4*4+3) * BPITCH + k] = (_Float16)bR[it].w;
      }
    } else {
      #pragma unroll
      for (int it = 0; it < BV; it++) {
        int idx = tid + it * 256;
        int k4 = idx & 7, n = idx >> 3;
        union { _Float16 h[4]; unsigned long long q; } u;
        u.h[0] = (_Float16)bR[it].x; u.h[1] = (_Float16)bR[it].y;
        u.h[2] = (_Float16)bR[it].z; u.h[3] = (_Float16)bR[it].w;
        *(unsigned long long*)(Bs + n * BPITCH + (k4 << 2)) = u.q;
      }
    }
    __syncthreads();

    // ---- fragments ----
    // A 16x32 f16: lanes 0-15 rows m, halves0-7 K=0..7, halves8-15 K=16..23
    //              lanes16-31 rows m, halves0-7 K=8..15, halves8-15 K=24..31
    const int koff = (lane & 16) ? 8 : 0;
    const int rl   = lane & 15;
    v16h afrag[2], bfrag[NI];
    #pragma unroll
    for (int mi = 0; mi < 2; mi++) {
      const _Float16* p = As + (wm * 32 + mi * 16 + rl) * APITCH;
      union { v16h v; v8h h[2]; } u;
      u.h[0] = *(const v8h*)(p + koff);
      u.h[1] = *(const v8h*)(p + 16 + koff);
      afrag[mi] = u.v;
    }
    // B 32x16 f16: lanes 0-15 col n, halves0-7 K=0..7, halves8-15 K=16..23; lanes16-31 K=8..15, 24..31
    const int kb = (lane & 16) ? 8 : 0;
    #pragma unroll
    for (int ni = 0; ni < NI; ni++) {
      const _Float16* p = Bs + (wn * (TBN/2) + ni * 16 + rl) * BPITCH + kb;
      union { v16h v; v8h h[2]; } u;
      u.h[0] = *(const v8h*)(p);
      u.h[1] = *(const v8h*)(p + 16);
      bfrag[ni] = u.v;
    }
    #pragma unroll
    for (int mi = 0; mi < 2; mi++)
      #pragma unroll
      for (int ni = 0; ni < NI; ni++)
        acc[mi][ni] = __builtin_amdgcn_wmma_f32_16x16x32_f16(
            false, afrag[mi], false, bfrag[ni], (short)0, acc[mi][ni], false, false);
  }

  // ---- epilogue: D layout lanes0-15 -> M=j,N=lane; lanes16-31 -> M=8+j,N=lane-16 ----
  const int nloc  = lane & 15;
  const int mhalf = (lane & 16) ? 8 : 0;
  #pragma unroll
  for (int mi = 0; mi < 2; mi++)
    #pragma unroll
    for (int ni = 0; ni < NI; ni++) {
      #pragma unroll
      for (int j = 0; j < 8; j++) {
        int gm = m0 + wm * 32 + mi * 16 + mhalf + j;
        int gn = n0 + wn * (TBN/2) + ni * 16 + nloc;
        if (gm < M && gn < N) {
          float vsum = acc[mi][ni][j];
          if (bias) vsum += bias_per_m ? bias[gm] : bias[gn];
          if (xres) {  // gather residual from x[B,C,T,V]; gm is token index
            int vv = gm % V_;
            int t  = (gm / V_) % T_;
            int b  = gm / (T_ * V_);
            vsum += xres[(((long)b * C_ + gn) * T_ + t) * V_ + vv];
          }
          if (act == 1)      vsum = vsum > 0.f ? vsum : 0.f;
          else if (act == 2) vsum = 1.f / (1.f + expf(-vsum));
          Out[(long)gm * ldo + gn] = vsum;
        }
      }
    }
}

// ---------------- small kernels ----------------

// xm[b,c,v] = mean_t x[b,c,t,v]
__global__ void k_mean_t(const float* __restrict__ x, float* __restrict__ xm) {
  long i = (long)blockIdx.x * 256 + threadIdx.x;
  if (i >= (long)B_ * C_ * V_) return;
  int v = i % V_; long r = i / V_;
  int c = r % C_; int b = r / C_;
  const float* p = x + (((long)b * C_ + c) * T_) * V_ + v;
  float s = 0.f;
  #pragma unroll 8
  for (int t = 0; t < T_; t++) s += p[(long)t * V_];
  xm[i] = s * (1.f / T_);
}

// x1[b,s,r,v] = conv1_w[s,r,:] . xm[b,:,v] + conv1_b[s,r]
__global__ void k_conv1(const float* __restrict__ w1, const float* __restrict__ b1,
                        const float* __restrict__ xm, float* __restrict__ x1) {
  long i = (long)blockIdx.x * 256 + threadIdx.x;
  if (i >= (long)B_ * S_ * REL_ * V_) return;
  int v = i % V_; long r = i / V_;
  int rr = r % REL_; r /= REL_;
  int s = r % S_; int b = r / S_;
  const float* wr = w1 + ((long)s * REL_ + rr) * C_;
  const float* xb = xm + ((long)b * C_) * V_ + v;
  float acc = b1[s * REL_ + rr];
  #pragma unroll 8
  for (int c = 0; c < C_; c++) acc += wr[c] * xb[(long)c * V_];
  x1[i] = acc;
}

// d[b,s,r,u,v] = tanh(x1[b,s,r,u] - x1[b,s,r,v])
__global__ void k_dtanh(const float* __restrict__ x1, float* __restrict__ dbuf) {
  long i = (long)blockIdx.x * 256 + threadIdx.x;
  if (i >= (long)B_ * S_ * REL_ * VV) return;
  int v = i % V_; long q = i / V_;
  int u = q % V_; long bsr = q / V_;
  const float* xr = x1 + bsr * V_;
  dbuf[i] = tanhf(xr[u] - xr[v]);
}

// adj[b,s,o,uv] = alpha*(conv4_w[s,o,:] . d[b,s,:,uv] + conv4_b[s,o]) + PA[s,uv]
__global__ void k_adj(const float* __restrict__ dbuf, const float* __restrict__ w4,
                      const float* __restrict__ b4, const float* __restrict__ PA,
                      const float* __restrict__ alpha_p, float* __restrict__ adj) {
  long i = (long)blockIdx.x * 256 + threadIdx.x;
  if (i >= (long)B_ * S_ * C_ * VV) return;
  int uv = i % VV; long r = i / VV;
  int o = r % C_; r /= C_;
  int s = r % S_; int b = r / S_;
  const float* dd = dbuf + ((long)(b * S_ + s) * REL_) * VV + uv;
  const float* wr = w4 + ((long)s * C_ + o) * REL_;
  float acc = 0.f;
  #pragma unroll 8
  for (int rr = 0; rr < REL_; rr++) acc += wr[rr] * dd[(long)rr * VV];
  adj[i] = alpha_p[0] * (acc + b4[s * C_ + o]) + PA[(long)s * VV + uv];
}

// x3m[b,s,t,v] = mean_o x3[b,s,o,t,v]
__global__ void k_x3m(const float* __restrict__ x3, float* __restrict__ x3m) {
  long i = (long)blockIdx.x * 256 + threadIdx.x;
  if (i >= (long)B_ * S_ * T_ * V_) return;
  int tv = i % TV;
  long bs = i / TV;
  const float* p = x3 + (bs * C_) * TV + tv;
  float s = 0.f;
  #pragma unroll 8
  for (int o = 0; o < C_; o++) s += p[(long)o * TV];
  x3m[i] = s * (1.f / C_);
}

// x4[b,s,t,m,n] = tanh(x3m[b,s,t,m] - x3m[b,s,t,n])
__global__ void k_x4(const float* __restrict__ x3m, float* __restrict__ x4) {
  long i = (long)blockIdx.x * 256 + threadIdx.x;
  if (i >= (long)B_ * S_ * T_ * VV) return;
  int n = i % V_; long q = i / V_;
  int m = q % V_; q /= V_;
  int t = q % T_; long bs = q / T_;
  const float* p = x3m + bs * TV + (long)t * V_;
  x4[i] = tanhf(p[m] - p[n]);
}

// local path mix: y = sum_s beta*(x3 @ adj^T) + gamma*(x4 . x3); BN; relu(y+x) -> comb[:, 0:256]
__global__ void __launch_bounds__(256) k_local_mix(
    const float* __restrict__ x, const float* __restrict__ adj,
    const float* __restrict__ x3, const float* __restrict__ x4,
    const float* __restrict__ bn_w, const float* __restrict__ bn_b,
    const float* __restrict__ beta_p, const float* __restrict__ gamma_p,
    float* __restrict__ comb)
{
  int bo = blockIdx.x;             // b*C_ + o
  int b = bo / C_, o = bo % C_;
  int tid = threadIdx.x;
  __shared__ __align__(16) float adj_s[VV];
  __shared__ __align__(16) float x3_s[TV];
  const float beta = beta_p[0], gamma = gamma_p[0];
  float acc[7];
  #pragma unroll
  for (int r = 0; r < 7; r++) acc[r] = 0.f;

  for (int s = 0; s < S_; s++) {
    long base = (long)(b * S_ + s) * C_ + o;
    const float* adj_g = adj + base * VV;
    const float* x3_g  = x3 + base * TV;
    __syncthreads();
    for (int i = tid; i < VV; i += 256) adj_s[i] = adj_g[i];
    // x3 tile: 400 16B chunks via async global->LDS DMA
    for (int i = tid; i < TV / 4; i += 256)
      async_copy16(&x3_s[i * 4], x3_g + i * 4);
    async_wait0();
    __syncthreads();
    const float* x4_g = x4 + (long)(b * S_ + s) * T_ * VV;
    #pragma unroll
    for (int r = 0; r < 7; r++) {
      int idx = tid + r * 256;
      if (idx < TV) {
        int t = idx / V_, u = idx % V_;
        const float* x3t  = x3_s + t * V_;
        const float* adju = adj_s + u * V_;
        const float* x4t  = x4_g + (long)t * VV + u * V_;
        float a1 = 0.f, a2 = 0.f;
        #pragma unroll
        for (int v = 0; v < V_; v++) {
          a1 += adju[v] * x3t[v];
          a2 += x4t[v] * x3t[v];
        }
        acc[r] += beta * a1 + gamma * a2;
      }
    }
  }
  const float scale = bn_w[o] * 0.99999500003750f;  // 1/sqrt(1+1e-5)
  const float bb = bn_b[o];
  #pragma unroll
  for (int r = 0; r < 7; r++) {
    int idx = tid + r * 256;
    if (idx < TV) {
      int t = idx / V_, u = idx % V_;
      float y  = acc[r] * scale + bb;
      float xv = x[(((long)b * C_ + o) * T_ + t) * V_ + u];
      float loc = y + xv;
      loc = loc > 0.f ? loc : 0.f;
      long token = (long)(b * T_ + t) * V_ + u;
      comb[token * 512 + o] = loc;
    }
  }
}

// LayerNorm over C of x (gathered from [B,C,T,V]) -> xn[token][c]; one wave per token
__global__ void k_ln1(const float* __restrict__ x, const float* __restrict__ w,
                      const float* __restrict__ bb, float* __restrict__ xn) {
  int wid = threadIdx.x >> 5, lane = threadIdx.x & 31;
  long token = (long)blockIdx.x * 8 + wid;
  if (token >= NT) return;
  int b = token / (T_ * V_);
  int rem = token % (T_ * V_);
  int t = rem / V_, v = rem % V_;
  const float* xb = x + (((long)b * C_) * T_ + t) * V_ + v;  // index c via stride TV
  float vals[8];
  float sum = 0.f;
  #pragma unroll
  for (int i = 0; i < 8; i++) { int c = lane + i * 32; vals[i] = xb[(long)c * TV]; sum += vals[i]; }
  for (int off = 16; off > 0; off >>= 1) sum += __shfl_xor(sum, off, 32);
  float mean = sum * (1.f / C_);
  float vs = 0.f;
  #pragma unroll
  for (int i = 0; i < 8; i++) { float d = vals[i] - mean; vs += d * d; }
  for (int off = 16; off > 0; off >>= 1) vs += __shfl_xor(vs, off, 32);
  float inv = rsqrtf(vs * (1.f / C_) + 1e-5f);
  #pragma unroll
  for (int i = 0; i < 8; i++) {
    int c = lane + i * 32;
    xn[token * C_ + c] = (vals[i] - mean) * inv * w[c] + bb[c];
  }
}

// attention per (n,h): scores = q.k + rpe[h,hop]; softmax(scores*0.125); o = attn@v
__global__ void __launch_bounds__(128) k_attn(
    const float* __restrict__ qkv, const float* __restrict__ rpe,
    const int* __restrict__ hop, float* __restrict__ out)
{
  int nh = blockIdx.x;
  int n = nh / H_, h = nh % H_;
  __shared__ __align__(16) float qs[V_ * HD_], ks[V_ * HD_], vs[V_ * HD_];
  __shared__ float sc[VV];
  int tid = threadIdx.x;
  // q/k/v tiles: 400 16B chunks each via async global->LDS DMA
  for (int ch = tid; ch < (V_ * HD_) / 4; ch += 128) {
    int iv = ch / (HD_ / 4), d4 = ch % (HD_ / 4);
    long base = (long)(n * V_ + iv) * (3 * C_) + h * HD_ + d4 * 4;
    int lo = iv * HD_ + d4 * 4;
    async_copy16(&qs[lo], qkv + base);
    async_copy16(&ks[lo], qkv + base + C_);
    async_copy16(&vs[lo], qkv + base + 2 * C_);
  }
  async_wait0();
  __syncthreads();
  for (int i = tid; i < VV; i += 128) {
    int ii = i / V_, jj = i % V_;
    float a = 0.f;
    #pragma unroll 8
    for (int d = 0; d < HD_; d++) a += qs[ii * HD_ + d] * ks[jj * HD_ + d];
    a += rpe[h * HOP1 + hop[i]];
    sc[i] = a * 0.125f;   // hd^-0.5
  }
  __syncthreads();
  if (tid < V_) {
    float mx = -1e30f;
    for (int j = 0; j < V_; j++) mx = fmaxf(mx, sc[tid * V_ + j]);
    float ssum = 0.f;
    for (int j = 0; j < V_; j++) { float e = expf(sc[tid * V_ + j] - mx); sc[tid * V_ + j] = e; ssum += e; }
    float r = 1.f / ssum;
    for (int j = 0; j < V_; j++) sc[tid * V_ + j] *= r;
  }
  __syncthreads();
  for (int i = tid; i < V_ * HD_; i += 128) {
    int ii = i / HD_, d = i % HD_;
    float a = 0.f;
    #pragma unroll
    for (int j = 0; j < V_; j++) a += sc[ii * V_ + j] * vs[j * HD_ + d];
    out[(long)(n * V_ + ii) * C_ + h * HD_ + d] = a;
  }
}

// LayerNorm over HID=64 in place; one wave per token
__global__ void k_gln(float* __restrict__ h, const float* __restrict__ w,
                      const float* __restrict__ bb) {
  int wid = threadIdx.x >> 5, lane = threadIdx.x & 31;
  long token = (long)blockIdx.x * 8 + wid;
  if (token >= NT) return;
  float* hp = h + token * HID_;
  float v0 = hp[lane], v1 = hp[lane + 32];
  float sum = v0 + v1;
  for (int off = 16; off > 0; off >>= 1) sum += __shfl_xor(sum, off, 32);
  float mean = sum * (1.f / HID_);
  float d0 = v0 - mean, d1 = v1 - mean;
  float vs = d0 * d0 + d1 * d1;
  for (int off = 16; off > 0; off >>= 1) vs += __shfl_xor(vs, off, 32);
  float inv = rsqrtf(vs * (1.f / HID_) + 1e-5f);
  hp[lane]      = d0 * inv * w[lane]      + bb[lane];
  hp[lane + 32] = d1 * inv * w[lane + 32] + bb[lane + 32];
}

// fused = gate*local + (1-gate)*glob, written in [B,C,T,V]
__global__ void k_fuse(const float* __restrict__ comb, const float* __restrict__ gate,
                       float* __restrict__ out) {
  long i = (long)blockIdx.x * 256 + threadIdx.x;
  if (i >= (long)NOUT) return;
  int vv = i % V_; long r = i / V_;
  int t = r % T_; r /= T_;
  int c = r % C_; int b = r / C_;
  long token = (long)(b * T_ + t) * V_ + vv;
  float g  = gate[token * C_ + c];
  float l  = comb[token * 512 + c];
  float gl = comb[token * 512 + 256 + c];
  out[i] = g * l + (1.f - g) * gl;
}

// ---------------- launch ----------------
static inline unsigned cdiv(long a, long b) { return (unsigned)((a + b - 1) / b); }

extern "C" void kernel_launch(void* const* d_in, const int* in_sizes, int n_in,
                              void* d_out, int out_size, void* d_ws, size_t ws_size,
                              hipStream_t stream) {
  (void)in_sizes; (void)n_in; (void)out_size; (void)ws_size;
  const float* x       = (const float*)d_in[0];
  const float* PA      = (const float*)d_in[1];
  const float* alpha_p = (const float*)d_in[2];
  const float* beta_p  = (const float*)d_in[3];
  const float* gamma_p = (const float*)d_in[4];
  const float* conv1_w = (const float*)d_in[5];
  const float* conv1_b = (const float*)d_in[6];
  const float* conv3_w = (const float*)d_in[7];
  const float* conv3_b = (const float*)d_in[8];
  const float* conv4_w = (const float*)d_in[9];
  const float* conv4_b = (const float*)d_in[10];
  const float* bn_w    = (const float*)d_in[11];
  const float* bn_b    = (const float*)d_in[12];
  const float* ln1_w   = (const float*)d_in[13];
  const float* ln1_b   = (const float*)d_in[14];
  const float* qkv_w   = (const float*)d_in[15];
  const float* out_w   = (const float*)d_in[16];
  const float* out_b   = (const float*)d_in[17];
  const float* rpe     = (const float*)d_in[18];
  const float* g1_w    = (const float*)d_in[19];
  const float* g1_b    = (const float*)d_in[20];
  const float* gln_w   = (const float*)d_in[21];
  const float* gln_b   = (const float*)d_in[22];
  const float* g2_w    = (const float*)d_in[23];
  const float* g2_b    = (const float*)d_in[24];
  const int*   hop     = (const int*)d_in[25];
  float* outp = (float*)d_out;

  // ---- workspace regions (floats), region-reused across phases; ~752 MB total ----
  float* ws   = (float*)d_ws;
  float* comb = ws;                            // 52,428,800  (persist: [token][512])
  float* R1   = ws + 52428800L;                // 78,643,200  (x3 -> qkv)
  float* R2   = R1 + 78643200L;                // 30,720,000  (adj -> xn -> h)
  float* R3   = R2 + 30720000L;                // 26,214,400  (x4/x3m/xm/x1/d -> attn_o -> gate)
  float* x3b   = R1;
  float* qkvb  = R1;
  float* adj   = R2;
  float* xn    = R2;
  float* hbuf  = R2;
  float* x4b   = R3;
  float* x3mb  = R3 + 7680000L;
  float* xmb   = R3 + 7987200L;
  float* x1b   = R3 + 8396800L;
  float* dbuf  = R3 + 8550400L;                // 3,840,000
  float* attno = R3;
  float* gate  = R3;

  dim3 blk256(256), blk128(128);

  // ===== local path =====
  k_mean_t<<<cdiv((long)B_*C_*V_, 256), blk256, 0, stream>>>(x, xmb);
  k_conv1 <<<cdiv((long)B_*S_*REL_*V_, 256), blk256, 0, stream>>>(conv1_w, conv1_b, xmb, x1b);
  k_dtanh <<<cdiv((long)B_*S_*REL_*VV, 256), blk256, 0, stream>>>(x1b, dbuf);
  k_adj   <<<cdiv((long)B_*S_*C_*VV, 256), blk256, 0, stream>>>(dbuf, conv4_w, conv4_b, PA, alpha_p, adj);

  // x3[b,s] = conv3_w[s] @ x[b]  (M=256, N=1600, K=256), bias per-m
  k_wmma_linear<64><<<dim3(C_/128, TV/64, B_*S_), blk256, 0, stream>>>(
      conv3_w, C_, x, TV, /*b_k_major=*/1, conv3_b, /*bias_per_m=*/1, nullptr,
      x3b, TV, C_, TV, C_, /*act=*/0,
      /*batch=*/1, (long)C_*C_, (long)C_*T_*V_, (long)C_*TV, (long)C_);

  k_x3m<<<cdiv((long)B_*S_*T_*V_, 256), blk256, 0, stream>>>(x3b, x3mb);
  k_x4 <<<cdiv((long)B_*S_*T_*VV, 256), blk256, 0, stream>>>(x3mb, x4b);
  k_local_mix<<<B_*C_, blk256, 0, stream>>>(x, adj, x3b, x4b, bn_w, bn_b, beta_p, gamma_p, comb);

  // ===== global path =====
  k_ln1<<<NT/8, blk256, 0, stream>>>(x, ln1_w, ln1_b, xn);

  // qkv = xn @ qkv_w^T  (M=102400, N=768, K=256)
  k_wmma_linear<128><<<dim3(NT/128, (3*C_)/128, 1), blk256, 0, stream>>>(
      xn, C_, qkv_w, C_, 0, nullptr, 0, nullptr,
      qkvb, 3*C_, NT, 3*C_, C_, 0, 0, 0, 0, 0, 0);

  k_attn<<<(B_*T_)*H_, blk128, 0, stream>>>(qkvb, rpe, hop, attno);

  // glob = xg + attno @ out_w^T + out_b  -> comb[:, 256:512]
  k_wmma_linear<128><<<dim3(NT/128, C_/128, 1), blk256, 0, stream>>>(
      attno, C_, out_w, C_, 0, out_b, 0, /*xres=*/x,
      comb + 256, 512, NT, C_, C_, 0, 0, 0, 0, 0, 0);

  // ===== gate fusion =====
  // h = relu(comb @ g1_w^T + g1_b)  (K=512, N=64)
  k_wmma_linear<64><<<dim3(NT/128, 1, 1), blk256, 0, stream>>>(
      comb, 512, g1_w, 512, 0, g1_b, 0, nullptr,
      hbuf, HID_, NT, HID_, 512, /*relu*/1, 0, 0, 0, 0, 0);

  k_gln<<<NT/8, blk256, 0, stream>>>(hbuf, gln_w, gln_b);

  // gate = sigmoid(h @ g2_w^T + g2_b)  (K=64, N=256)
  k_wmma_linear<128><<<dim3(NT/128, C_/128, 1), blk256, 0, stream>>>(
      hbuf, HID_, g2_w, HID_, 0, g2_b, 0, nullptr,
      gate, C_, NT, C_, HID_, /*sigmoid*/2, 0, 0, 0, 0, 0);

  k_fuse<<<cdiv((long)NOUT, 256), blk256, 0, stream>>>(comb, gate, outp);
}
